// AdaptiveMultiWIRE_81741817577964
// MI455X (gfx1250) — compile-verified, matrix operand order checked by reading, not run on hardware
//
#include <hip/hip_runtime.h>

// AdaptiveMultiWIRE fused kernel for gfx1250 (MI455X).
// C=16 channels, N=8192 points, H=181 hidden, OUT=3, NIN=2, NB=8.
// All layers fused; activations live in LDS as f16 WMMA A-fragments.

#define C_      16
#define N_      8192
#define H_      181
#define OUT_    3
#define NIN_    2
#define NB_     8
#define HP_     192            // H padded to multiple of 32
#define KC_N    6              // 192/32 K-chunks
#define CT_N    12             // 192/16 output column tiles
#define ROWS    64             // rows per workgroup
#define WAVES   4
#define THREADS 128
#define FRAG_BYTES 1024        // 32 lanes * 16 f16 * 2B
#define XBYTES  (2*2*4*KC_N*FRAG_BYTES)   // 2 bufs * {re,im} * 4 row-tiles * 6 chunks = 98304
#define BBYTES  (4*KC_N*FRAG_BYTES)       // 4 weight planes * 6 chunks = 24576
#define SMEM_BYTES (XBYTES + BBYTES)      // 122880

#define OMEGA_  30.0f
#define S2_     100.0f         // SCALE^2

typedef __attribute__((ext_vector_type(16))) _Float16 v16h;
typedef __attribute__((ext_vector_type(8)))  float    v8f;

#define WMMA_F16(A,B,Cacc) \
  __builtin_amdgcn_wmma_f32_16x16x32_f16(false,(A),false,(B),(short)0,(Cacc),false,false)

// byte offset of a fragment in the X activation area
__device__ __forceinline__ unsigned xbase(int buf, int mat, int mt, int kc) {
  return (unsigned)(((((buf * 2 + mat) * 4 + mt) * KC_N) + kc) * FRAG_BYTES);
}
// byte offset of a fragment in the B staging area (w: 0=Wa_r 1=Wa_i 2=Wb_r 3=Wb_i)
__device__ __forceinline__ unsigned bbase(int w, int kc) {
  return (unsigned)(XBYTES + (w * KC_N + kc) * FRAG_BYTES);
}
// Store one f16 element into a fragment at matrix coords (row 0..15, k_local 0..31)
// using the documented 16-bit A/B VGPR layout:
//   k = (e&7) + (e>=8 ? 16 : 0) + (lane>=16 ? 8 : 0);  row = lane & 15
__device__ __forceinline__ void st_h(char* s, unsigned base, int row, int k, float v) {
  int e  = (k & 7) | ((k >> 1) & 8);
  int ln = row | ((k & 8) << 1);
  *reinterpret_cast<_Float16*>(s + base + (unsigned)((ln * 16 + e) * 2)) = (_Float16)v;
}

__global__ void __launch_bounds__(THREADS)
wire_fused_kernel(const float* __restrict__ inp, const int* __restrict__ indices,
                  const int* __restrict__ model_idx, const int* __restrict__ bias_idx,
                  const float* __restrict__ W0a, const float* __restrict__ b0a,
                  const float* __restrict__ W0b, const float* __restrict__ b0b,
                  const float* __restrict__ W1a, const float* __restrict__ b1a,
                  const float* __restrict__ W1b, const float* __restrict__ b1b,
                  const float* __restrict__ W2a, const float* __restrict__ b2a,
                  const float* __restrict__ W2b, const float* __restrict__ b2b,
                  const float* __restrict__ Wf,  const float* __restrict__ bf,
                  float* __restrict__ out) {
  extern __shared__ char smem[];
  const int tid  = threadIdx.x;
  const int wave = tid >> 5;
  const int lane = tid & 31;
  const int c    = blockIdx.y;
  const int row0 = blockIdx.x * ROWS;
  const int src  = indices[c];
  const int mi   = model_idx[c];
  const int bi   = bias_idx[c];

  // Zero both activation buffers (padded K lanes must be finite/zero for WMMA).
  for (int i = tid; i < XBYTES / 4; i += THREADS) ((unsigned*)smem)[i] = 0u;
  __syncthreads();

  // ---------------- layer 0: real 2->181 linear + Gabor, write f16 A-fragments ----
  {
    const int rbase = row0 + wave * 16;
    for (int t = lane; t < 16 * H_; t += 32) {
      int m = t / H_;
      int o = t - m * H_;
      float x0 = inp[(size_t)(src * N_ + rbase + m) * NIN_ + 0];
      float x1 = inp[(size_t)(src * N_ + rbase + m) * NIN_ + 1];
      float la = x0 * W0a[(mi * NIN_ + 0) * H_ + o] + x1 * W0a[(mi * NIN_ + 1) * H_ + o]
               + b0a[bi * H_ + o];
      float lb = x0 * W0b[(mi * NIN_ + 0) * H_ + o] + x1 * W0b[(mi * NIN_ + 1) * H_ + o]
               + b0b[bi * H_ + o];
      float g = __expf(-S2_ * (la * la + lb * lb));
      float sn, cs;
      __sincosf(OMEGA_ * la, &sn, &cs);
      st_h(smem, xbase(0, 0, wave, o >> 5), m, o & 31, g * cs);
      st_h(smem, xbase(0, 1, wave, o >> 5), m, o & 31, g * sn);
    }
  }
  // X row-tiles are wave-private: no barrier needed for X itself.

  // ---------------- hidden complex layers 1 and 2 (WMMA) ---------------------------
  for (int layer = 0; layer < 2; ++layer) {
    const float* Wa = layer ? W2a : W1a;
    const float* Wb = layer ? W2b : W1b;
    const float* ba = layer ? b2a : b1a;
    const float* bb = layer ? b2b : b1b;
    const int bin  = layer & 1;   // layer1: read buf0 write buf1; layer2: read buf1 write buf0
    const int bout = bin ^ 1;

    for (int ct = 0; ct < CT_N; ++ct) {
      const int o0 = ct * 16;
      __syncthreads();  // previous readers of B done
      for (int i = tid; i < BBYTES / 4; i += THREADS)
        ((unsigned*)(smem + XBYTES))[i] = 0u;   // zero pads (k>=181, o>=181)
      __syncthreads();
      // Stage this column tile's weights as f16 B-fragments: 4 planes x 6 chunks.
      for (int t = tid; t < 2 * H_ * 8; t += THREADS) {
        int tens = t / (H_ * 8);
        int rem  = t - tens * (H_ * 8);
        int k    = rem >> 3;
        int part = rem & 7;
        const float* Wsrc = tens ? Wb : Wa;
        const float* gp = Wsrc + ((size_t)(mi * H_ + k) * H_ + o0) * 2 + part * 4;
#pragma unroll
        for (int j = 0; j < 4; ++j) {
          int off = part * 4 + j;      // (o_local, ri) packed: contiguous f32 within row
          int ol  = off >> 1;
          int ri  = off & 1;
          if (o0 + ol < H_) {
            st_h(smem, bbase(tens * 2 + ri, k >> 5), ol, k & 31, gp[j]);
          }
        }
      }
      __syncthreads();

      // Each wave: its 16-row tile, this 16-col tile, full K. Complex GEMM pair.
      v8f lap = {}, laq = {}, lai = {}, lbp = {}, lbq = {}, lbi = {};
#pragma unroll
      for (int kc = 0; kc < KC_N; ++kc) {
        v16h Ar  = *(const v16h*)(smem + xbase(bin, 0, wave, kc) + (unsigned)(lane * 32));
        v16h Ai  = *(const v16h*)(smem + xbase(bin, 1, wave, kc) + (unsigned)(lane * 32));
        v16h Bar = *(const v16h*)(smem + bbase(0, kc) + (unsigned)(lane * 32));
        v16h Bai = *(const v16h*)(smem + bbase(1, kc) + (unsigned)(lane * 32));
        v16h Bbr = *(const v16h*)(smem + bbase(2, kc) + (unsigned)(lane * 32));
        v16h Bbi = *(const v16h*)(smem + bbase(3, kc) + (unsigned)(lane * 32));
        lap = WMMA_F16(Ar, Bar, lap);   // Re(la) = Xr*Wr - Xi*Wi  (P - Q)
        laq = WMMA_F16(Ai, Bai, laq);
        lai = WMMA_F16(Ar, Bai, lai);   // Im(la) = Xr*Wi + Xi*Wr
        lai = WMMA_F16(Ai, Bar, lai);
        lbp = WMMA_F16(Ar, Bbr, lbp);
        lbq = WMMA_F16(Ai, Bbi, lbq);
        lbi = WMMA_F16(Ar, Bbi, lbi);
        lbi = WMMA_F16(Ai, Bbr, lbi);
      }

      // bias + Gabor activation, write next layer's A-fragments (wave-private rows)
      int o = o0 + (lane & 15);
      float bar_ = 0.f, bai_ = 0.f, bbr_ = 0.f, bbi_ = 0.f;
      if (o < H_) {
        bar_ = ba[(bi * H_ + o) * 2 + 0];
        bai_ = ba[(bi * H_ + o) * 2 + 1];
        bbr_ = bb[(bi * H_ + o) * 2 + 0];
        bbi_ = bb[(bi * H_ + o) * 2 + 1];
      }
#pragma unroll
      for (int j = 0; j < 8; ++j) {
        float lar = lap[j] - laq[j] + bar_;
        float lim = lai[j] + bai_;
        float lbr = lbp[j] - lbq[j] + bbr_;
        float lbm = lbi[j] + bbi_;
        float mag = S2_ * (lar * lar + lim * lim + lbr * lbr + lbm * lbm);
        float g = __expf(-OMEGA_ * lim - mag);
        float sn, cs;
        __sincosf(OMEGA_ * lar, &sn, &cs);
        int m = j + ((lane >> 4) << 3);  // C/D layout: lanes>=16 hold rows 8..15
        st_h(smem, xbase(bout, 0, wave, o >> 5), m, o & 31, g * cs);
        st_h(smem, xbase(bout, 1, wave, o >> 5), m, o & 31, g * sn);
      }
    }
  }

  // ---------------- final complex 181->3 layer, keep real part --------------------
  __syncthreads();
  for (int i = tid; i < BBYTES / 4; i += THREADS) ((unsigned*)(smem + XBYTES))[i] = 0u;
  __syncthreads();
  for (int t = tid; t < H_ * OUT_ * 2; t += THREADS) {
    int k   = t / (OUT_ * 2);
    int rem = t - k * (OUT_ * 2);
    int oo  = rem >> 1;
    int ri  = rem & 1;
    float v = Wf[((size_t)(mi * H_ + k) * OUT_ + oo) * 2 + ri];
    st_h(smem, bbase(ri, k >> 5), oo, k & 31, v);
  }
  __syncthreads();

  v8f p = {}, q = {};
#pragma unroll
  for (int kc = 0; kc < KC_N; ++kc) {
    v16h Ar = *(const v16h*)(smem + xbase(0, 0, wave, kc) + (unsigned)(lane * 32));
    v16h Ai = *(const v16h*)(smem + xbase(0, 1, wave, kc) + (unsigned)(lane * 32));
    v16h Br = *(const v16h*)(smem + bbase(0, kc) + (unsigned)(lane * 32));
    v16h Bi = *(const v16h*)(smem + bbase(1, kc) + (unsigned)(lane * 32));
    p = WMMA_F16(Ar, Br, p);   // Re(out) = Xr*Wfr - Xi*Wfi
    q = WMMA_F16(Ai, Bi, q);
  }
  int oo = lane & 15;
  if (oo < OUT_) {
    float bfr = bf[(bi * OUT_ + oo) * 2 + 0];
#pragma unroll
    for (int j = 0; j < 8; ++j) {
      int row = row0 + wave * 16 + j + ((lane >> 4) << 3);
      out[((size_t)c * N_ + row) * OUT_ + oo] = p[j] - q[j] + bfr;
    }
  }
}

extern "C" void kernel_launch(void* const* d_in, const int* in_sizes, int n_in,
                              void* d_out, int out_size, void* d_ws, size_t ws_size,
                              hipStream_t stream) {
  (void)in_sizes; (void)n_in; (void)d_ws; (void)ws_size; (void)out_size;
  const float* inp       = (const float*)d_in[0];
  const int*   indices   = (const int*)d_in[1];
  const int*   model_idx = (const int*)d_in[2];
  const int*   bias_idx  = (const int*)d_in[3];
  const float* W0a = (const float*)d_in[4];
  const float* b0a = (const float*)d_in[5];
  const float* W0b = (const float*)d_in[6];
  const float* b0b = (const float*)d_in[7];
  const float* W1a = (const float*)d_in[8];
  const float* b1a = (const float*)d_in[9];
  const float* W1b = (const float*)d_in[10];
  const float* b1b = (const float*)d_in[11];
  const float* W2a = (const float*)d_in[12];
  const float* b2a = (const float*)d_in[13];
  const float* W2b = (const float*)d_in[14];
  const float* b2b = (const float*)d_in[15];
  const float* Wf  = (const float*)d_in[16];
  const float* bf  = (const float*)d_in[17];
  float* outp = (float*)d_out;

  // 120 KB dynamic LDS (> default 64 KB): raise the limit every call (deterministic,
  // not a stream op, safe under graph capture).
  hipFuncSetAttribute((const void*)wire_fused_kernel,
                      hipFuncAttributeMaxDynamicSharedMemorySize, SMEM_BYTES);

  dim3 grid(N_ / ROWS, C_);   // 128 row-tiles x 16 channels
  dim3 block(THREADS);
  wire_fused_kernel<<<grid, block, SMEM_BYTES, stream>>>(
      inp, indices, model_idx, bias_idx,
      W0a, b0a, W0b, b0b,
      W1a, b1a, W1b, b1b,
      W2a, b2a, W2b, b2b,
      Wf, bf, outp);
}
